// Experts_26508538151133
// MI455X (gfx1250) — compile-verified, matrix operand order checked by reading, not run on hardware
//
#include <hip/hip_runtime.h>
#include <math.h>

// ---------------------------------------------------------------------------
// Grouped experts MLP:  out[b,e] = gelu(x[b,e] @ w1[e] + b1[e]) @ w2[e] + b2[e]
// B=2 E=8 N=1024 D=1024 H=4096, fp32 in/out.
//
// MI455X strategy:
//   * compute-bound (~600 FLOP/byte) -> bf16 WMMA (v_wmma_f32_16x16x32_bf16)
//   * pay fp32->bf16 conversion + weight transpose ONCE in streaming prep
//     kernels (weights are reused by every tile; 192MB L2 holds an expert)
//   * GEMM inner loop: pure 16B async copies global->LDS
//     (global_load_async_to_lds_b128, ASYNCcnt) with double-buffered LDS,
//     one barrier per K-step, WMMA overlapped with prefetch.
// Workspace layout (bytes):
//   xb  [B,E,N,D]  bf16 @ 0          (32 MB)
//   w1t [E,H,D]    bf16 @ 32MB       (64 MB)   (transposed: row n, cols k)
//   w2t [E,D,H]    bf16 @ 96MB       (64 MB)
//   h   [B,E,N,H]  bf16 @ 160MB      (128 MB)
// ---------------------------------------------------------------------------

typedef __bf16 bf16;
typedef bf16  v16bf __attribute__((ext_vector_type(16)));
typedef bf16  v8bf  __attribute__((ext_vector_type(8)));
typedef float v8f   __attribute__((ext_vector_type(8)));
typedef int   v4i   __attribute__((vector_size(16)));

#define AS1 __attribute__((address_space(1)))
#define AS3 __attribute__((address_space(3)))

#define B_  2
#define E_  8
#define N_  1024
#define D_  1024
#define H_  4096

#define TM  128
#define TN  256
#define TK  32
#define LDT 40   // padded LDS row stride (bf16): 20 dwords -> conflict-free rows

// ---------------- CDNA5 async global->LDS copy (ASYNCcnt) -------------------
__device__ __forceinline__ void async_cp16(const bf16* g, bf16* l) {
#if __has_builtin(__builtin_amdgcn_global_load_async_to_lds_b128)
    __builtin_amdgcn_global_load_async_to_lds_b128(
        (AS1 v4i*)g, (AS3 v4i*)l, 0, 0);
#else
    unsigned loff = (unsigned)(unsigned long long)(AS3 const bf16*)l;
    asm volatile("global_load_async_to_lds_b128 %0, %1, off"
                 :: "v"(loff), "v"(g) : "memory");
#endif
}

__device__ __forceinline__ void wait_async0() {
#if __has_builtin(__builtin_amdgcn_s_wait_asynccnt)
    __builtin_amdgcn_s_wait_asynccnt(0);
#else
    asm volatile("s_wait_asynccnt 0x0" ::: "memory");
#endif
}

// ---------------- helpers ---------------------------------------------------
__device__ __forceinline__ v16bf frag_from_lds(const bf16* p) {
    // lanes 0-15: K=[0..7]&[16..23]; lanes 16-31: K=[8..15]&[24..31]
    // (caller bakes koff = (lane>>4)*8 into p)
    v8bf lo = *(const v8bf*)(p);
    v8bf hi = *(const v8bf*)(p + 16);
    v16bf f;
#pragma unroll
    for (int i = 0; i < 8; ++i) { f[i] = lo[i]; f[i + 8] = hi[i]; }
    return f;
}

__device__ __forceinline__ float gelu_erf(float x) {
    return 0.5f * x * (1.0f + erff(x * 0.70710678118654752f));
}

// ---------------- prep: elementwise fp32 -> bf16 ---------------------------
__global__ __launch_bounds__(256)
void convert_bf16_kernel(const float* __restrict__ src, bf16* __restrict__ dst) {
    const size_t g = (size_t)blockIdx.x * 256 + threadIdx.x;
    const float4* p = (const float4*)src + g * 2;
    float4 f0 = p[0], f1 = p[1];
    v8bf v;
    v[0] = (bf16)f0.x; v[1] = (bf16)f0.y; v[2] = (bf16)f0.z; v[3] = (bf16)f0.w;
    v[4] = (bf16)f1.x; v[5] = (bf16)f1.y; v[6] = (bf16)f1.z; v[7] = (bf16)f1.w;
    *(v8bf*)(dst + g * 8) = v;
}

// ---------------- prep: per-expert transpose + fp32 -> bf16 ----------------
// src: [E][R][C] f32   ->   dst: [E][C][R] bf16.   grid = (C/64, R/64, E)
__global__ __launch_bounds__(256)
void transpose_convert_kernel(const float* __restrict__ src, bf16* __restrict__ dst,
                              int R, int C) {
    __shared__ bf16 sT[64 * 72];
    const float* s = src + (size_t)blockIdx.z * R * C;
    bf16*        d = dst + (size_t)blockIdx.z * R * C;
    const int c0 = blockIdx.x * 64;
    const int r0 = blockIdx.y * 64;
    const int t  = threadIdx.x;

    // load 64x64 fp32 tile, convert, stage in LDS
    {
        const int rr = t >> 2;
        const int cc = (t & 3) * 16;
        const float4* p = (const float4*)(s + (size_t)(r0 + rr) * C + c0 + cc);
        bf16* q = &sT[rr * 72 + cc];
#pragma unroll
        for (int j = 0; j < 4; ++j) {
            float4 f = p[j];
            q[j * 4 + 0] = (bf16)f.x; q[j * 4 + 1] = (bf16)f.y;
            q[j * 4 + 2] = (bf16)f.z; q[j * 4 + 3] = (bf16)f.w;
        }
    }
    __syncthreads();
    // write transposed: dst[c][r], coalesced 32B per thread
    {
        const int cw = t >> 2;
        const int kc = (t & 3) * 16;
        v8bf o0, o1;
#pragma unroll
        for (int i = 0; i < 8; ++i) o0[i] = sT[(kc + i) * 72 + cw];
#pragma unroll
        for (int i = 0; i < 8; ++i) o1[i] = sT[(kc + 8 + i) * 72 + cw];
        bf16* q = d + (size_t)(c0 + cw) * R + r0 + kc;
        *(v8bf*)(q)     = o0;
        *(v8bf*)(q + 8) = o1;
    }
}

// ---------------- main GEMM: C = act(A @ Bt^T + bias) ----------------------
// A  : bf16 [N_][K]      row-major (lda = K)
// Bt : bf16 [NOUT][K]    row-major (pre-transposed weights, ld = K)
// 128x256 tile / block of 8 wave32; each wave: 64x64 = 4x4 WMMA frags.
// Double-buffered LDS fed by async b128 copies; one barrier per K-step.
template<int K, int NOUT, bool GELU>
__global__ __launch_bounds__(256)
void gemm_tile(const bf16* __restrict__ A, const bf16* __restrict__ Bt,
               const float* __restrict__ bias, void* __restrict__ Cout)
{
    __shared__ bf16 sA[2][TM * LDT];   // 2 x 10 KB
    __shared__ bf16 sB[2][TN * LDT];   // 2 x 20 KB

    const int be = blockIdx.y;
    const int e  = be & (E_ - 1);
    const int nTiles = NOUT / TN;
    const int mt = blockIdx.x / nTiles;
    const int nt = blockIdx.x % nTiles;
    const int m0 = mt * TM;
    const int n0 = nt * TN;

    const bf16*  Ab = A    + (size_t)be * N_ * K;
    const bf16*  Bb = Bt   + (size_t)e  * NOUT * K;
    const float* bb = bias + (size_t)e  * NOUT;

    const int tid   = threadIdx.x;
    const int lane  = tid & 31;
    const int wave  = tid >> 5;
    const int waveM = wave >> 2;                // 0..1 -> 64 rows
    const int waveN = wave & 3;                 // 0..3 -> 64 cols

    const int lr = tid >> 2;                    // 0..63  loader row
    const int ak = (tid & 3) * 8;               // 0,8,16,24 k-chunk

    v8f acc[4][4];
#pragma unroll
    for (int i = 0; i < 4; ++i) {
#pragma unroll
        for (int j = 0; j < 4; ++j) acc[i][j] = (v8f)(0.0f);
    }

    const int koff = (lane >> 4) * 8;
    const int rsel = lane & 15;
    const int KT = K / TK;

    // stage(k-step 0) into buffer 0
    {
#pragma unroll
        for (int rr = 0; rr < 2; ++rr) {
            const int r = lr + rr * 64;
            async_cp16(Ab + (size_t)(m0 + r) * K + ak, &sA[0][r * LDT + ak]);
        }
#pragma unroll
        for (int rr = 0; rr < 4; ++rr) {
            const int n = lr + rr * 64;
            async_cp16(Bb + (size_t)(n0 + n) * K + ak, &sB[0][n * LDT + ak]);
        }
    }

    for (int kt = 0; kt < KT; ++kt) {
        const int bi = kt & 1;
        wait_async0();          // my async copies into buf[bi] are complete
        __syncthreads();        // everyone's copies done; buf[bi^1] free

        if (kt + 1 < KT) {      // prefetch next K-step into the other buffer
            const int k0 = (kt + 1) * TK;
#pragma unroll
            for (int rr = 0; rr < 2; ++rr) {
                const int r = lr + rr * 64;
                async_cp16(Ab + (size_t)(m0 + r) * K + k0 + ak,
                           &sA[bi ^ 1][r * LDT + ak]);
            }
#pragma unroll
            for (int rr = 0; rr < 4; ++rr) {
                const int n = lr + rr * 64;
                async_cp16(Bb + (size_t)(n0 + n) * K + k0 + ak,
                           &sB[bi ^ 1][n * LDT + ak]);
            }
        }

        v16bf afr[4];
#pragma unroll
        for (int mi = 0; mi < 4; ++mi)
            afr[mi] = frag_from_lds(&sA[bi][(waveM * 64 + mi * 16 + rsel) * LDT + koff]);
        v16bf bfr[4];
#pragma unroll
        for (int ni = 0; ni < 4; ++ni)
            bfr[ni] = frag_from_lds(&sB[bi][(waveN * 64 + ni * 16 + rsel) * LDT + koff]);

#pragma unroll
        for (int mi = 0; mi < 4; ++mi) {
#pragma unroll
            for (int ni = 0; ni < 4; ++ni) {
                acc[mi][ni] = __builtin_amdgcn_wmma_f32_16x16x32_bf16(
                    false, afr[mi], false, bfr[ni], (short)0, acc[mi][ni],
                    false, false);
            }
        }
    }

    // ---- epilogue ----
    const int rbase = m0 + waveM * 64 + ((lane >> 4) << 3);
    const int cbase = n0 + waveN * 64 + (lane & 15);
#pragma unroll
    for (int ni = 0; ni < 4; ++ni) {
        const int col  = cbase + ni * 16;
        const float bv = bb[col];
#pragma unroll
        for (int mi = 0; mi < 4; ++mi) {
            const int r0v = rbase + mi * 16;
#pragma unroll
            for (int v = 0; v < 8; ++v) {
                const float val = acc[mi][ni][v] + bv;
                if (GELU) {
                    bf16* C = (bf16*)Cout + (size_t)be * N_ * NOUT;
                    C[(size_t)(r0v + v) * NOUT + col] = (bf16)gelu_erf(val);
                } else {
                    float* C = (float*)Cout + (size_t)be * N_ * NOUT;
                    C[(size_t)(r0v + v) * NOUT + col] = val;
                }
            }
        }
    }
}

// ---------------------------------------------------------------------------
extern "C" void kernel_launch(void* const* d_in, const int* in_sizes, int n_in,
                              void* d_out, int out_size, void* d_ws, size_t ws_size,
                              hipStream_t stream) {
    (void)in_sizes; (void)n_in; (void)out_size; (void)ws_size;
    const float* x  = (const float*)d_in[0];   // [B,E,N,D]
    const float* w1 = (const float*)d_in[1];   // [E,D,H]
    const float* b1 = (const float*)d_in[2];   // [E,H]
    const float* w2 = (const float*)d_in[3];   // [E,H,D]
    const float* b2 = (const float*)d_in[4];   // [E,D]
    float* out = (float*)d_out;                // [B,E,N,D]

    char* ws = (char*)d_ws;
    const size_t nX  = (size_t)B_ * E_ * N_ * D_;   // 16.7M
    const size_t nW1 = (size_t)E_ * D_ * H_;        // 33.6M
    const size_t nW2 = (size_t)E_ * H_ * D_;        // 33.6M
    bf16* xb  = (bf16*)(ws);
    bf16* w1t = (bf16*)(ws + nX * 2);
    bf16* w2t = (bf16*)(ws + (nX + nW1) * 2);
    bf16* h   = (bf16*)(ws + (nX + nW1 + nW2) * 2); // [B,E,N,H] bf16

    dim3 blk(256);

    // prep: convert x; transpose+convert w1 ([E,D,H]->[E,H,D]) and w2
    convert_bf16_kernel<<<dim3((unsigned)(nX / (256 * 8))), blk, 0, stream>>>(x, xb);
    transpose_convert_kernel<<<dim3(H_ / 64, D_ / 64, E_), blk, 0, stream>>>(w1, w1t, D_, H_);
    transpose_convert_kernel<<<dim3(D_ / 64, H_ / 64, E_), blk, 0, stream>>>(w2, w2t, H_, D_);

    // pass 1: h = gelu(xb @ w1 + b1)   (bf16 out)
    dim3 grid1((N_ / TM) * (H_ / TN), B_ * E_);   // (8*16, 16)
    gemm_tile<D_, H_, true><<<grid1, blk, 0, stream>>>(xb, w1t, b1, (void*)h);

    // pass 2: out = h @ w2 + b2        (fp32 out)
    dim3 grid2((N_ / TM) * (D_ / TN), B_ * E_);   // (8*4, 16)
    gemm_tile<H_, D_, false><<<grid2, blk, 0, stream>>>(h, w2t, b2, (void*)out);
}
